// CorrelationModule_4672924418157
// MI455X (gfx1250) — matrix operand order repaired and checked by reading, not compiled, not run on hardware
//
#include <hip/hip_runtime.h>
#include <math.h>

#define B_DIM 4
#define C_DIM 64
#define HW    64
#define NPIX  4096
#define EPS   1e-5f

typedef __attribute__((ext_vector_type(16))) _Float16 v16h;
typedef __attribute__((ext_vector_type(8)))  _Float16 v8h;
typedef __attribute__((ext_vector_type(2)))  _Float16 v2h;
typedef __attribute__((ext_vector_type(8)))  float    v8f;

union HU { v16h h; unsigned u[8]; };

static __device__ __forceinline__ v8f wmma_f16(v16h a, v16h b, v8f c) {
  // D = A(16x32 f16) * B(32x16 f16) + C(16x16 f32)
  return __builtin_amdgcn_wmma_f32_16x16x32_f16(false, a, false, b, (short)0, c,
                                                false, false);
}

// A-operand (16x32 f16) from row-major f16 memory (row stride ld halves).
// lanes 0-15: row=lane, halves 0..15 = K0..15;
// lanes 16-31: same row, K16..31.
static __device__ __forceinline__ v16h load_a_tile(const _Float16* base, int ld, int lane) {
  const int r  = lane & 15;
  const int hi = (lane >> 4) & 1;
  const _Float16* p = base + r * ld + hi * 16;
  v8h lo = *(const v8h*)(p);
  v8h hh = *(const v8h*)(p + 8);
  v16h out;
#pragma unroll
  for (int i = 0; i < 8; ++i) { out[i] = lo[i]; out[i + 8] = hh[i]; }
  return out;
}

// B-operand (32x16 f16, KxN): memory holds element [n][k] contiguous in k.
// lanes 0-15: col=lane, K0..15; lanes 16-31: col=lane-16, K16..31.
static __device__ __forceinline__ v16h load_b_tile(const _Float16* base, int ld, int lane) {
  const int n    = lane & 15;
  const int koff = (lane >> 4) * 16;
  const _Float16* p = base + n * ld + koff;
  v8h b0 = *(const v8h*)(p);
  v8h b1 = *(const v8h*)(p + 8);
  v16h out;
#pragma unroll
  for (int i = 0; i < 8; ++i) { out[i] = b0[i]; out[i + 8] = b1[i]; }
  return out;
}

// ---------------- prep: ex_corr = ex^T @ W_lin^T  -> f16 [B][N][C] ----------------
__global__ __launch_bounds__(256) void linear_kernel(const float* __restrict__ ex,
                                                     const float* __restrict__ Wl,
                                                     _Float16* __restrict__ exc) {
  __shared__ float wl[C_DIM * C_DIM];
  const int tid = threadIdx.x;
  for (int i = tid; i < C_DIM * C_DIM; i += 256) wl[i] = Wl[i];
  __syncthreads();

  const int b = blockIdx.y;
  const int n = blockIdx.x * 256 + tid;
  const float* exB = ex + ((size_t)b * C_DIM) * NPIX + n;

  float xv[C_DIM];
#pragma unroll
  for (int k = 0; k < C_DIM; ++k) xv[k] = exB[(size_t)k * NPIX];

  _Float16* out = exc + ((size_t)b * NPIX + n) * C_DIM;
#pragma unroll 4
  for (int c = 0; c < C_DIM; ++c) {
    float acc = 0.f;
#pragma unroll
    for (int k = 0; k < C_DIM; ++k) acc += wl[c * C_DIM + k] * xv[k];
    out[c] = (_Float16)acc;
  }
}

// ---------------- prep: fp16 copies of q in [C][N] and transposed [N][C] ----------
__global__ __launch_bounds__(256) void pack_q_kernel(const float* __restrict__ q,
                                                     _Float16* __restrict__ qV,
                                                     _Float16* __restrict__ qT) {
  const size_t i = (size_t)blockIdx.x * 256 + threadIdx.x;  // over B*C*N
  const float v = q[i];
  qV[i] = (_Float16)v;
  const int    n  = (int)(i & (NPIX - 1));
  const size_t bc = i >> 12;
  const int    c  = (int)(bc & (C_DIM - 1));
  const size_t b  = bc >> 6;
  qT[((b * NPIX + n) * C_DIM) + c] = (_Float16)v;
}

// ---------------- flash attention (S transposed, 32 queries per wave) -------------
// S^T[n,m] = sum_c qT[n,c]*exc[m,c]  -> A = qT rows (n), B = exc (loop-invariant).
// C-layout: lane = column m, VGPR index = row n  => softmax axis is in-lane.
// O[c,m]  = sum_n qV[c,n]*P[m,n]    -> A = qV rows (c), B = P^T from registers.
// Two query groups g=0,1 (m0+g*16 .. +15) share every loaded aK / aV tile:
// 32 WMMAs per 32 b128 loads per 64-key chunk (2x the intensity of 16 q/wave).
#define WAVES 2
__global__ __launch_bounds__(32 * WAVES) void attn_kernel(const _Float16* __restrict__ exc,
                                                          const _Float16* __restrict__ qT,
                                                          const _Float16* __restrict__ qV,
                                                          float* __restrict__ att) {
  const int lane = threadIdx.x & 31;
  const int wave = threadIdx.x >> 5;
  const int tile = blockIdx.x * WAVES + wave;      // 0 .. B*(N/32)-1
  const int b    = tile >> 7;                      // 128 query-tiles per batch
  const int m0   = (tile & 127) * 32;
  const int h    = lane >> 4;                      // lane half

  const _Float16* excB = exc + (size_t)b * NPIX * C_DIM;
  const _Float16* qTB  = qT  + (size_t)b * NPIX * C_DIM;
  const _Float16* qVB  = qV  + (size_t)b * C_DIM * NPIX;

  // Loop-invariant B operands: ex_corr rows, two groups of 16 queries.
  v16h bE[2][2];
#pragma unroll
  for (int g = 0; g < 2; ++g) {
    bE[g][0] = load_b_tile(excB + (size_t)(m0 + g * 16) * C_DIM,      C_DIM, lane);
    bE[g][1] = load_b_tile(excB + (size_t)(m0 + g * 16) * C_DIM + 32, C_DIM, lane);
  }

  v8f o[2][4];
#pragma unroll
  for (int g = 0; g < 2; ++g)
#pragma unroll
    for (int ci = 0; ci < 4; ++ci) o[g][ci] = (v8f){};
  float rm[2] = { -INFINITY, -INFINITY };
  float rl[2] = { 0.f, 0.f };

  for (int n0 = 0; n0 < NPIX; n0 += 64) {
    // ---- S^T: each aK tile pair feeds both query groups ----
    v8f s[2][4];
#pragma unroll
    for (int j = 0; j < 4; ++j) {
      const v16h aK0 = load_a_tile(qTB + (size_t)(n0 + j * 16) * C_DIM,      C_DIM, lane);
      const v16h aK1 = load_a_tile(qTB + (size_t)(n0 + j * 16) * C_DIM + 32, C_DIM, lane);
#pragma unroll
      for (int g = 0; g < 2; ++g) {
        v8f t = (v8f){};
        t = wmma_f16(aK0, bE[g][0], t);
        t = wmma_f16(aK1, bE[g][1], t);
        s[g][j] = t;
      }
    }

    // ---- online softmax per group (in-lane reduce + one partner exchange) ----
    v16h bP[2][2];
#pragma unroll
    for (int g = 0; g < 2; ++g) {
      float ml = s[g][0][0];
#pragma unroll
      for (int j = 0; j < 4; ++j)
#pragma unroll
        for (int i = 0; i < 8; ++i) ml = fmaxf(ml, s[g][j][i]);
      ml = fmaxf(ml, __shfl_xor(ml, 16, 32));

      const float nm = fmaxf(rm[g], ml);
      const float sc = __expf(rm[g] - nm);
      rm[g] = nm;
#pragma unroll
      for (int ci = 0; ci < 4; ++ci)
#pragma unroll
        for (int i = 0; i < 8; ++i) o[g][ci][i] = o[g][ci][i] * sc;

      float sl = 0.f;
#pragma unroll
      for (int j = 0; j < 4; ++j)
#pragma unroll
        for (int i = 0; i < 8; ++i) {
          const float e = __expf(s[g][j][i] - nm);
          s[g][j][i] = e;
          sl += e;
        }
      rl[g] = rl[g] * sc + sl + __shfl_xor(sl, 16, 32);

      // pack P to f16 pairs; this lane owns (j, h) blocks: n = j*16+h*8+{0..7}
      unsigned pk[4][4];
#pragma unroll
      for (int j = 0; j < 4; ++j)
#pragma unroll
        for (int p = 0; p < 4; ++p) {
          v2h pr = { (_Float16)s[g][j][2 * p], (_Float16)s[g][j][2 * p + 1] };
          pk[j][p] = __builtin_bit_cast(unsigned, pr);
        }

      // assemble B-operand tiles of P^T (K = n): one half-lane swap per tile
#pragma unroll
      for (int t = 0; t < 2; ++t) {
        unsigned recv[4];
#pragma unroll
        for (int d = 0; d < 4; ++d) {
          const unsigned snd = h ? pk[2 * t][d] : pk[2 * t + 1][d];
          recv[d] = __shfl_xor(snd, 16, 32);
        }
        HU u;
#pragma unroll
        for (int d = 0; d < 4; ++d) {
          u.u[d]     = h ? recv[d]          : pk[2 * t][d];     // (j=2t+h, h'=0)
          u.u[4 + d] = h ? pk[2 * t + 1][d] : recv[d];          // (j=2t+h, h'=1)
        }
        bP[g][t] = u.h;
      }
    }

    // ---- O += qV x P^T : each aV tile pair feeds both query groups ----
#pragma unroll
    for (int ci = 0; ci < 4; ++ci) {
      const v16h aV0 = load_a_tile(qVB + (size_t)(ci * 16) * NPIX + n0,      NPIX, lane);
      const v16h aV1 = load_a_tile(qVB + (size_t)(ci * 16) * NPIX + n0 + 32, NPIX, lane);
#pragma unroll
      for (int g = 0; g < 2; ++g) {
        o[g][ci] = wmma_f16(aV0, bP[g][0], o[g][ci]);
        o[g][ci] = wmma_f16(aV1, bP[g][1], o[g][ci]);
      }
    }
  }

  // ---- epilogue: 1/l scale, store att in natural [B][C][N] layout (coalesced) ----
  float* outB = att + (size_t)b * C_DIM * NPIX;
#pragma unroll
  for (int g = 0; g < 2; ++g) {
    const float inv = 1.0f / rl[g];
#pragma unroll
    for (int ci = 0; ci < 4; ++ci)
#pragma unroll
      for (int i = 0; i < 8; ++i) {
        const int c = ci * 16 + i + 8 * h;
        outB[(size_t)c * NPIX + m0 + g * 16 + (lane & 15)] = o[g][ci][i] * inv;
      }
  }
}

// ---------------- conv3x3 + BN(eval) + LeakyReLU(0.1), att in [C][N] --------------
__global__ __launch_bounds__(64) void conv_bn_kernel(const float* __restrict__ att,
                                                     const float* __restrict__ cw,
                                                     const float* __restrict__ gamma,
                                                     const float* __restrict__ beta,
                                                     const float* __restrict__ mean,
                                                     const float* __restrict__ var,
                                                     float* __restrict__ out) {
  __shared__ float wl[C_DIM * 9];
  const int w   = threadIdx.x;
  const int hh0 = blockIdx.x;
  const int co  = blockIdx.y;
  const int b   = blockIdx.z;

  const float* wsrc = cw + (size_t)co * C_DIM * 9;   // OIHW: [co][ci][ky][kx]
  for (int i = w; i < C_DIM * 9; i += 64) wl[i] = wsrc[i];
  __syncthreads();

  const float* aB = att + (size_t)b * C_DIM * NPIX;
  float acc = 0.f;
#pragma unroll 8
  for (int ci = 0; ci < C_DIM; ++ci) {
    const float* ap = aB + (size_t)ci * NPIX;
    const float* wp = wl + ci * 9;
#pragma unroll
    for (int ky = 0; ky < 3; ++ky) {
      const int hy = hh0 + ky - 1;
      if ((unsigned)hy >= HW) continue;
#pragma unroll
      for (int kx = 0; kx < 3; ++kx) {
        const int wx = w + kx - 1;
        if ((unsigned)wx >= HW) continue;
        acc += wp[ky * 3 + kx] * ap[hy * HW + wx];   // lane-coalesced loads
      }
    }
  }

  const float is = rsqrtf(var[co] + EPS);
  float y = (acc - mean[co]) * (gamma[co] * is) + beta[co];
  y = (y > 0.f) ? y : 0.1f * y;
  out[((size_t)(b * C_DIM + co)) * NPIX + hh0 * HW + w] = y;
}

// ---------------------------------------------------------------------------------
extern "C" void kernel_launch(void* const* d_in, const int* in_sizes, int n_in,
                              void* d_out, int out_size, void* d_ws, size_t ws_size,
                              hipStream_t stream) {
  const float* ex    = (const float*)d_in[0];
  const float* q     = (const float*)d_in[1];
  const float* Wl    = (const float*)d_in[2];
  const float* cw    = (const float*)d_in[3];
  const float* gamma = (const float*)d_in[4];
  const float* beta  = (const float*)d_in[5];
  const float* mean  = (const float*)d_in[6];
  const float* var   = (const float*)d_in[7];
  float* out = (float*)d_out;

  const size_t halfBuf = (size_t)B_DIM * NPIX * C_DIM * sizeof(_Float16); // 2 MB each
  char* ws = (char*)d_ws;
  _Float16* exc = (_Float16*)(ws);
  _Float16* qT  = (_Float16*)(ws + halfBuf);
  _Float16* qV  = (_Float16*)(ws + 2 * halfBuf);
  float*    att = (float*)   (ws + 3 * halfBuf);   // 4 MB f32 [B][C][N]

  linear_kernel<<<dim3(NPIX / 256, B_DIM), 256, 0, stream>>>(ex, Wl, exc);
  pack_q_kernel<<<dim3((B_DIM * C_DIM * NPIX) / 256), 256, 0, stream>>>(q, qV, qT);
  attn_kernel<<<dim3((B_DIM * NPIX / 32) / WAVES), 32 * WAVES, 0, stream>>>(exc, qT, qV, att);
  conv_bn_kernel<<<dim3(HW, C_DIM, B_DIM), 64, 0, stream>>>(att, cw, gamma, beta, mean, var, out);
}
